// LorenzPINN_88519275971190
// MI455X (gfx1250) — matrix operand order repaired and checked by reading, not compiled, not run on hardware
//
#include <hip/hip_runtime.h>

typedef __attribute__((ext_vector_type(16))) _Float16 v16h;
typedef __attribute__((ext_vector_type(8)))  float    v8f;
typedef __attribute__((ext_vector_type(8)))  int      v8i;

#define WMMA16x16x32(A, B, C) \
  __builtin_amdgcn_wmma_f32_16x16x32_f16(false, (A), false, (B), (short)0, (C), false, false)

__device__ __forceinline__ float tanh_fast(float x) {
#if __has_builtin(__builtin_amdgcn_tanhf)
  return __builtin_amdgcn_tanhf(x);  // gfx1250 v_tanh_f32: single trans op
#else
  // tanh(x) = 1 - 2/(exp2(x*2*log2e)+1): mul + v_exp + add + v_rcp + fma
  float e = __builtin_amdgcn_exp2f(x * 2.8853900817779268f);
  return 1.0f - 2.0f * __builtin_amdgcn_rcpf(e + 1.0f);
#endif
}

// v_cvt_pk_rtz_f16_f32: two f32 -> one packed-f16 dword
__device__ __forceinline__ int pk16(float a, float b) {
  return __builtin_bit_cast(int, __builtin_amdgcn_cvt_pkrtz(a, b));
}

// xor-16 half-wave swap as pure VALU: v_permlanex16_b32 with identity lane selects
__device__ __forceinline__ int swap16(int x) {
  return __builtin_amdgcn_permlanex16(x, x, 0x76543210, 0xfedcba98, false, false);
}

// Build one 16x32 f16 A-tile holding W^T rows j = jbase..jbase+15, K = k (0..19 valid),
// with K==20 carrying the bias (bias folded into the matmul via a constant-1 row in B).
// A layout (16-bit, 16x32): lane<16 -> M=lane, halves h: K=h;
//                           lane>=16 -> M=lane-16, halves h: K=h+16.
__device__ __forceinline__ v16h build_wT_tile(const float* __restrict__ W,
                                              const float* __restrict__ b,
                                              int ld, int jmax, int jbase, int lane) {
  const int  j  = jbase + (lane & 15);
  const bool hi = lane >= 16;
  v16h a = {};
  if (j < jmax) {
#pragma unroll
    for (int h = 0; h < 16; ++h) {
      const int k = hi ? h + 16 : h;
      float v = 0.0f;
      if (k < 20)       v = W[k * ld + j];
      else if (k == 20) v = b[j];
      a[h] = (_Float16)v;
    }
  }
  return a;
}

// One hidden layer (20->20, tanh) for value (Bh) and tangent (Bd), in-place.
// D layout: lane<16 -> (j=vgpr, batch=lane); lane>=16 -> (j=vgpr+8, batch=lane-16).
// Next B layout: lane<16 -> word w = K pair (2w,2w+1); lane>=16 -> K pair (16+2w,17+2w).
// Only an xor-16 swap of packed-f16 words is needed; K==20 slot = f16(1.0) / 0.
__device__ __forceinline__ void mlp_layer(const v16h A1, const v16h A2,
                                          v16h& Bh, v16h& Bd, bool lo) {
  v8f z = {};
  v8f Dv1 = WMMA16x16x32(A1, Bh, z);
  v8f Dv2 = WMMA16x16x32(A2, Bh, z);
  v8f Dt1 = WMMA16x16x32(A1, Bd, z);
  v8f Dt2 = WMMA16x16x32(A2, Bd, z);

  // tanh + tangent scaling; tile2: only j=16..19 (i<4) are ever consumed.
  float hv1[8], dt1[8], hv2[4], dt2[4];
#pragma unroll
  for (int i = 0; i < 8; ++i) {
    float h = tanh_fast(Dv1[i]);
    hv1[i] = h;
    dt1[i] = (1.0f - h * h) * Dt1[i];
  }
#pragma unroll
  for (int i = 0; i < 4; ++i) {
    float h = tanh_fast(Dv2[i]);
    hv2[i] = h;
    dt2[i] = (1.0f - h * h) * Dt2[i];
  }

  // Pack adjacent-j pairs to f16 words (same source lane), then swap packed words.
  int pv1[4], pt1[4];
#pragma unroll
  for (int i = 0; i < 4; ++i) {
    pv1[i] = pk16(hv1[2 * i], hv1[2 * i + 1]);
    pt1[i] = pk16(dt1[2 * i], dt1[2 * i + 1]);
  }
  int pv2[2] = { pk16(hv2[0], hv2[1]), pk16(hv2[2], hv2[3]) };
  int pt2[2] = { pk16(dt2[0], dt2[1]), pk16(dt2[2], dt2[3]) };

  int sv1[4], st1[4];
#pragma unroll
  for (int i = 0; i < 4; ++i) { sv1[i] = swap16(pv1[i]); st1[i] = swap16(pt1[i]); }
  const int sv2[2] = { swap16(pv2[0]), swap16(pv2[1]) };
  const int st2[2] = { swap16(pt2[0]), swap16(pt2[1]) };

  // Assemble B: 8 word-level selects per matrix.
  v8i wh = {}, wd = {};
  wh[0] = lo ? pv1[0] : sv2[0];          // lo: K=0,1   hi: K=16,17
  wh[1] = lo ? pv1[1] : sv2[1];          // lo: K=2,3   hi: K=18,19
  wh[2] = lo ? pv1[2] : 0x00003C00;      // lo: K=4,5   hi: K=20 bias-input = f16(1.0)
  wh[3] = lo ? pv1[3] : 0;               // lo: K=6,7   hi: K=22,23 = 0
  wd[0] = lo ? pt1[0] : st2[0];
  wd[1] = lo ? pt1[1] : st2[1];
  wd[2] = lo ? pt1[2] : 0;               // no bias in tangent
  wd[3] = lo ? pt1[3] : 0;
#pragma unroll
  for (int i = 0; i < 4; ++i) {          // lo: K=8..15  hi: K=24..31 = 0
    wh[4 + i] = lo ? sv1[i] : 0;
    wd[4 + i] = lo ? st1[i] : 0;
  }
  Bh = __builtin_bit_cast(v16h, wh);
  Bd = __builtin_bit_cast(v16h, wd);
}

__global__ __launch_bounds__(256) void lorenz_pinn_wmma(
    const float* __restrict__ t,
    const float* __restrict__ W1, const float* __restrict__ b1,
    const float* __restrict__ W2, const float* __restrict__ b2,
    const float* __restrict__ W3, const float* __restrict__ b3,
    const float* __restrict__ W4, const float* __restrict__ b4,
    const float* __restrict__ Wo, const float* __restrict__ bo,
    const float* __restrict__ c1p, const float* __restrict__ c2p,
    const float* __restrict__ c3p,
    float* __restrict__ out, int n) {
  const int lane   = threadIdx.x & 31;
  const int wave   = blockIdx.x * (blockDim.x >> 5) + (threadIdx.x >> 5);
  const int nwaves = gridDim.x * (blockDim.x >> 5);
  const int ntiles = (n + 15) >> 4;
  const bool lo    = lane < 16;

  // Weight A-tiles: resident in VGPRs for the whole grid-stride loop (7 x 8 VGPRs).
  const v16h A2a = build_wT_tile(W2, b2, 20, 20, 0,  lane);
  const v16h A2b = build_wT_tile(W2, b2, 20, 20, 16, lane);
  const v16h A3a = build_wT_tile(W3, b3, 20, 20, 0,  lane);
  const v16h A3b = build_wT_tile(W3, b3, 20, 20, 16, lane);
  const v16h A4a = build_wT_tile(W4, b4, 20, 20, 0,  lane);
  const v16h A4b = build_wT_tile(W4, b4, 20, 20, 16, lane);
  const v16h Ao  = build_wT_tile(Wo, bo, 3,  3,  0,  lane);

  // Layer-1 weights per lane, hoisted out of the loop:
  // lane<16 handles K=i, lane>=16 handles K=16+i (K>=20 -> 0).
  float w1v[16], b1v[16];
#pragma unroll
  for (int i = 0; i < 16; ++i) {
    const int K = lo ? i : 16 + i;
    w1v[i] = (K < 20) ? W1[K] : 0.0f;
    b1v[i] = (K < 20) ? b1[K] : 0.0f;
  }

  const float c1 = c1p[0], c2 = c2p[0], c3 = c3p[0];

  for (int tile = wave; tile < ntiles; tile += nwaves) {
    const int base = tile << 4;
    int ti = base + (lane & 15);
    if (ti > n - 1) ti = n - 1;  // pad lanes recompute last point; never stored
    const float tb = t[ti];

    // Layer 1 (1->20) is rank-1: build value/tangent directly in B layout.
    float hv[16], dv[16];
#pragma unroll
    for (int i = 0; i < 16; ++i) {
      const float h = tanh_fast(tb * w1v[i] + b1v[i]);
      hv[i] = h;
      dv[i] = (1.0f - h * h) * w1v[i];  // dt = 1
    }
    hv[4] = lo ? hv[4] : 1.0f;  // hi lanes: K=20 bias-input row (dv[4] already 0)

    v8i wh0, wd0;
#pragma unroll
    for (int i = 0; i < 8; ++i) {
      wh0[i] = pk16(hv[2 * i], hv[2 * i + 1]);
      wd0[i] = pk16(dv[2 * i], dv[2 * i + 1]);
    }
    v16h Bh = __builtin_bit_cast(v16h, wh0);
    v16h Bd = __builtin_bit_cast(v16h, wd0);

    mlp_layer(A2a, A2b, Bh, Bd, lo);
    mlp_layer(A3a, A3b, Bh, Bd, lo);
    mlp_layer(A4a, A4b, Bh, Bd, lo);

    // Output layer 20->3 (bias via K=20 slot; tangent B has 0 there -> no bias in JVP).
    v8f z = {};
    v8f Ov = WMMA16x16x32(Ao, Bh, z);
    v8f Ot = WMMA16x16x32(Ao, Bd, z);

    if (lo) {
      const int row = base + lane;
      if (row < n) {
        const float x  = Ov[0], y  = Ov[1], zz = Ov[2];
        const float dx = Ot[0], dy = Ot[1], dz = Ot[2];
        const float fx = dx - c1 * (y - x);
        const float fy = dy - x * (c2 - zz) + y;
        const float fz = dz - x * y + c3 * zz;
        float2* o = reinterpret_cast<float2*>(out + (size_t)row * 6);  // 8B-aligned
        o[0] = make_float2(x, y);
        o[1] = make_float2(zz, fx);
        o[2] = make_float2(fy, fz);
      }
    }
  }
}

extern "C" void kernel_launch(void* const* d_in, const int* in_sizes, int n_in,
                              void* d_out, int out_size, void* d_ws, size_t ws_size,
                              hipStream_t stream) {
  const float* t  = (const float*)d_in[0];
  const float* W1 = (const float*)d_in[1];
  const float* b1 = (const float*)d_in[2];
  const float* W2 = (const float*)d_in[3];
  const float* b2 = (const float*)d_in[4];
  const float* W3 = (const float*)d_in[5];
  const float* b3 = (const float*)d_in[6];
  const float* W4 = (const float*)d_in[7];
  const float* b4 = (const float*)d_in[8];
  const float* Wo = (const float*)d_in[9];
  const float* bo = (const float*)d_in[10];
  const float* c1 = (const float*)d_in[11];
  const float* c2 = (const float*)d_in[12];
  const float* c3 = (const float*)d_in[13];
  float* out = (float*)d_out;
  const int n = in_sizes[0];

  // 1024 blocks x 8 waves = 8192 waves; 62500 tiles -> ~8 tiles/wave, amortizing
  // the per-wave weight-tile gather while saturating the WGPs.
  lorenz_pinn_wmma<<<1024, 256, 0, stream>>>(t, W1, b1, W2, b2, W3, b3, W4, b4,
                                             Wo, bo, c1, c2, c3, out, n);
}
